// GraphSAGE_8504035246140
// MI455X (gfx1250) — compile-verified, matrix operand order checked by reading, not run on hardware
//
#include <hip/hip_runtime.h>
#include <hip/hip_bf16.h>

typedef __attribute__((ext_vector_type(16))) __bf16 v16bf;
typedef __attribute__((ext_vector_type(8)))  float  v8f;

union BF16x16 {
    v16bf v;
    unsigned short u[16];
    uint4 q[2];
};

union BF16x8 {
    unsigned short u[8];
    uint4 q;
};

__device__ inline unsigned short f32_to_bf16(float f) {
    return __builtin_bit_cast(unsigned short, (__bf16)f);   // native RNE cvt
}

// Load one packed 16x32-bf16 operand tile (A or B layout identical once
// packed): 32 bytes per lane, contiguous -> two global_load_b128.
__device__ inline v16bf load_tile(const unsigned short* __restrict__ P,
                                  int slot, int lane) {
    BF16x16 r;
    const uint4* p = (const uint4*)(P + ((size_t)(slot * 32 + lane) * 16));
    r.q[0] = p[0];
    r.q[1] = p[1];
    return r.v;
}

// -------------------- small utility kernels --------------------
__global__ __launch_bounds__(256) void zero_kernel(float* __restrict__ p, int n) {
    int i = blockIdx.x * 256 + threadIdx.x;
    if (i < n) p[i] = 0.f;
}

__global__ __launch_bounds__(256) void deg_kernel(const int* __restrict__ dst,
                                                  float* __restrict__ deg, int E) {
    int e = blockIdx.x * 256 + threadIdx.x;
    if (e < E)
        __hip_atomic_fetch_add(&deg[dst[e]], 1.0f, __ATOMIC_RELAXED,
                               __HIP_MEMORY_SCOPE_AGENT);
}

__global__ __launch_bounds__(256) void deginv_kernel(float* __restrict__ deg, int N) {
    int i = blockIdx.x * 256 + threadIdx.x;
    if (i < N) deg[i] = 1.0f / fmaxf(deg[i], 1.0f);
}

// Pack W[K,128] (k-major row-major) into WMMA B-operand lane layout:
// flat = ((n_tile*(K/32)+kb)*32 + lane)*16 + j
__global__ __launch_bounds__(256) void pack_w_kernel(const float* __restrict__ W,
                                                     unsigned short* __restrict__ Wp,
                                                     int K) {
    int idx = blockIdx.x * 256 + threadIdx.x;
    if (idx >= K * 128) return;
    int j    = idx & 15;
    int lane = (idx >> 4) & 31;
    int slot = idx >> 9;
    int KB   = K >> 5;
    int n_tile = slot / KB;
    int kb     = slot - n_tile * KB;
    int half   = lane >> 4;
    int n = n_tile * 16 + (lane & 15);
    int k = kb * 32 + ((j < 8) ? (half * 8 + j) : (16 + half * 8 + (j - 8)));
    Wp[idx] = f32_to_bf16(W[k * 128 + n]);
}

// Pack X[M,128] (optionally scaled per-row by rowscale) into the WMMA
// 16x32 bf16 A-operand layout. One thread per 8-element group (16B store).
__global__ __launch_bounds__(256) void pack_a_kernel(const float* __restrict__ X,
                                                     const float* __restrict__ rowscale,
                                                     unsigned short* __restrict__ Ap,
                                                     int Mtiles) {
    int idx = blockIdx.x * 256 + threadIdx.x;
    if (idx >= Mtiles * 256) return;           // Mtiles*4slots*32lanes*2groups
    int g    = idx & 1;
    int lane = (idx >> 1) & 31;
    int slot = idx >> 6;                       // m_tile*4 + kb
    int kb   = slot & 3;
    int m_tile = slot >> 2;
    int half = lane >> 4;
    int row  = m_tile * 16 + (lane & 15);
    int k    = kb * 32 + half * 8 + g * 16;
    const float* p = X + (size_t)row * 128 + k;
    float4 f0 = *(const float4*)(p);
    float4 f1 = *(const float4*)(p + 4);
    float s = rowscale ? rowscale[row] : 1.0f;
    float t[8] = {f0.x, f0.y, f0.z, f0.w, f1.x, f1.y, f1.z, f1.w};
    BF16x8 o;
#pragma unroll
    for (int i = 0; i < 8; ++i) o.u[i] = f32_to_bf16(t[i] * s);
    *(uint4*)(Ap + ((size_t)slot * 32 + lane) * 16 + g * 8) = o.q;
}

// Pack pair features concat(h[ia], h[ib], |h[ia]-h[ib]|) [B,384] into the
// A-operand layout (KB=12). Gathers happen exactly once per element.
__global__ __launch_bounds__(256) void pack_pair_kernel(const float* __restrict__ h,
                                                        const int* __restrict__ ia,
                                                        const int* __restrict__ ib,
                                                        unsigned short* __restrict__ Ap,
                                                        int Mtiles) {
    int idx = blockIdx.x * 256 + threadIdx.x;
    if (idx >= Mtiles * 768) return;           // Mtiles*12slots*32lanes*2groups
    int g    = idx & 1;
    int lane = (idx >> 1) & 31;
    int slot = idx >> 6;                       // m_tile*12 + kb
    int kb   = slot % 12;
    int m_tile = slot / 12;
    int half = lane >> 4;
    int row  = m_tile * 16 + (lane & 15);
    int k    = kb * 32 + half * 8 + g * 16;    // 0..383
    int seg  = k >> 7;
    int kk   = k & 127;
    const float* pa = h + (size_t)ia[row] * 128 + kk;
    const float* pb = h + (size_t)ib[row] * 128 + kk;
    float t[8];
    if (seg == 0) {
        float4 f0 = *(const float4*)(pa), f1 = *(const float4*)(pa + 4);
        t[0]=f0.x; t[1]=f0.y; t[2]=f0.z; t[3]=f0.w;
        t[4]=f1.x; t[5]=f1.y; t[6]=f1.z; t[7]=f1.w;
    } else if (seg == 1) {
        float4 f0 = *(const float4*)(pb), f1 = *(const float4*)(pb + 4);
        t[0]=f0.x; t[1]=f0.y; t[2]=f0.z; t[3]=f0.w;
        t[4]=f1.x; t[5]=f1.y; t[6]=f1.z; t[7]=f1.w;
    } else {
        float4 a0 = *(const float4*)(pa), a1 = *(const float4*)(pa + 4);
        float4 b0 = *(const float4*)(pb), b1 = *(const float4*)(pb + 4);
        t[0]=fabsf(a0.x-b0.x); t[1]=fabsf(a0.y-b0.y);
        t[2]=fabsf(a0.z-b0.z); t[3]=fabsf(a0.w-b0.w);
        t[4]=fabsf(a1.x-b1.x); t[5]=fabsf(a1.y-b1.y);
        t[6]=fabsf(a1.z-b1.z); t[7]=fabsf(a1.w-b1.w);
    }
    BF16x8 o;
#pragma unroll
    for (int i = 0; i < 8; ++i) o.u[i] = f32_to_bf16(t[i]);
    *(uint4*)(Ap + ((size_t)slot * 32 + lane) * 16 + g * 8) = o.q;
}

// Fused gather + scatter-add: agg[dst] += h[src]. One wave per edge, fully
// coalesced 512B/edge; h is L2-resident so this runs at L2/atomic rate.
__global__ __launch_bounds__(256) void scatter_kernel(const float* __restrict__ h,
                                                      const int* __restrict__ src,
                                                      const int* __restrict__ dst,
                                                      float* __restrict__ agg, int E) {
    int gid = blockIdx.x * 256 + threadIdx.x;
    int e = gid >> 5;
    if (e >= E) return;
    int c = (gid & 31) * 4;
    float4 v = *(const float4*)(h + (size_t)src[e] * 128 + c);
    float* p = agg + (size_t)dst[e] * 128 + c;
    __hip_atomic_fetch_add(p + 0, v.x, __ATOMIC_RELAXED, __HIP_MEMORY_SCOPE_AGENT);
    __hip_atomic_fetch_add(p + 1, v.y, __ATOMIC_RELAXED, __HIP_MEMORY_SCOPE_AGENT);
    __hip_atomic_fetch_add(p + 2, v.z, __ATOMIC_RELAXED, __HIP_MEMORY_SCOPE_AGENT);
    __hip_atomic_fetch_add(p + 3, v.w, __ATOMIC_RELAXED, __HIP_MEMORY_SCOPE_AGENT);
}

// Generic strip-mined WMMA GEMM over pre-packed bf16 operands.
// Each wave owns MSTRIP m-tiles x 1 n-tile; B tiles are loaded once per kb
// and reused across the strip. DUAL accumulates two GEMMs (self+neigh).
template <int KB, int MSTRIP, bool DUAL, bool RELU>
__global__ __launch_bounds__(256) void wmma_gemm_kernel(
    const unsigned short* __restrict__ Ap, const unsigned short* __restrict__ Ap2,
    const unsigned short* __restrict__ Bp, const unsigned short* __restrict__ Bp2,
    const float* __restrict__ bias, float* __restrict__ out, int strips) {
    int lane = threadIdx.x & 31;
    int wave = threadIdx.x >> 5;
    int wid = blockIdx.x * 8 + wave;
    if (wid >= strips * 8) return;             // wave-uniform exit, EXEC stays full
    int strip = wid >> 3, n_tile = wid & 7;
    int m0 = strip * MSTRIP;
    v8f acc[MSTRIP] = {};
#pragma unroll
    for (int kb = 0; kb < KB; ++kb) {
        v16bf b1 = load_tile(Bp, n_tile * KB + kb, lane);
        v16bf b2;
        if constexpr (DUAL) b2 = load_tile(Bp2, n_tile * KB + kb, lane);
#pragma unroll
        for (int s = 0; s < MSTRIP; ++s) {
            v16bf a1 = load_tile(Ap, (m0 + s) * KB + kb, lane);
            acc[s] = __builtin_amdgcn_wmma_f32_16x16x32_bf16(
                false, a1, false, b1, (short)0, acc[s], false, false);
            if constexpr (DUAL) {
                v16bf a2 = load_tile(Ap2, (m0 + s) * KB + kb, lane);
                acc[s] = __builtin_amdgcn_wmma_f32_16x16x32_bf16(
                    false, a2, false, b2, (short)0, acc[s], false, false);
            }
        }
    }
    int half = lane >> 4;
    int col = n_tile * 16 + (lane & 15);
    float bs = bias[col];
#pragma unroll
    for (int s = 0; s < MSTRIP; ++s) {
#pragma unroll
        for (int r = 0; r < 8; ++r) {
            int row = (m0 + s) * 16 + r + 8 * half;   // C/D layout: VGPR r -> M=r/r+8
            float v = acc[s][r] + bs;
            if (RELU) v = fmaxf(v, 0.f);
            out[(size_t)row * 128 + col] = v;
        }
    }
}

// Per-column mean / rstd over the batch dim for both z buffers.
__global__ __launch_bounds__(256) void bn_stats_kernel(const float* __restrict__ zsrc,
                                                       const float* __restrict__ ztar,
                                                       float* __restrict__ stats,
                                                       int Brows) {
    __shared__ float ss[256], sq[256];
    int bid = blockIdx.x;                 // 0..255 : batch*128 + col
    int batch = bid >> 7, col = bid & 127;
    const float* zp = batch ? ztar : zsrc;
    float s = 0.f, s2 = 0.f;
    for (int r = threadIdx.x; r < Brows; r += 256) {
        float v = zp[(size_t)r * 128 + col];
        s += v; s2 += v * v;
    }
    ss[threadIdx.x] = s; sq[threadIdx.x] = s2;
    __syncthreads();
    for (int o = 128; o > 0; o >>= 1) {
        if ((int)threadIdx.x < o) {
            ss[threadIdx.x] += ss[threadIdx.x + o];
            sq[threadIdx.x] += sq[threadIdx.x + o];
        }
        __syncthreads();
    }
    if (threadIdx.x == 0) {
        float inv = 1.0f / (float)Brows;
        float m = ss[0] * inv;
        float var = sq[0] * inv - m * m;
        stats[batch * 128 + col]       = m;
        stats[256 + batch * 128 + col] = rsqrtf(var + 1e-5f);
    }
}

// Normalize + ReLU + store mmd outputs; fused tiny Dx2 GEMM (h_p) via
// wave32 shuffle reduction for the src batch.
__global__ __launch_bounds__(256) void finalize_kernel(
    const float* __restrict__ zsrc, const float* __restrict__ ztar,
    const float* __restrict__ stats, const float* __restrict__ gamma,
    const float* __restrict__ beta, const float* __restrict__ W2,
    const float* __restrict__ b2, float* __restrict__ out, int Brows) {
    int lane = threadIdx.x & 31;
    int wave = threadIdx.x >> 5;
    int w = blockIdx.x * 8 + wave;
    if (w >= 2 * Brows) return;
    int batch = (w >= Brows) ? 1 : 0;
    int row   = batch ? (w - Brows) : w;
    const float* zp   = batch ? ztar : zsrc;
    const float* mean = stats + batch * 128;
    const float* rstd = stats + 256 + batch * 128;
    float4 zv = *(const float4*)(zp + (size_t)row * 128 + lane * 4);
    float vv[4] = {zv.x, zv.y, zv.z, zv.w};
    float ov[4];
    float p0 = 0.f, p1 = 0.f;
#pragma unroll
    for (int j = 0; j < 4; ++j) {
        int d = lane * 4 + j;
        float v = (vv[j] - mean[d]) * rstd[d] * gamma[d] + beta[d];
        v = fmaxf(v, 0.f);
        ov[j] = v;
        p0 += v * W2[d * 2 + 0];
        p1 += v * W2[d * 2 + 1];
    }
    float4 o = {ov[0], ov[1], ov[2], ov[3]};
    size_t base = (size_t)2 * Brows + (batch ? (size_t)Brows * 128 : 0);
    *(float4*)(out + base + (size_t)row * 128 + lane * 4) = o;
    if (batch == 0) {
#pragma unroll
        for (int off = 16; off > 0; off >>= 1) {
            p0 += __shfl_xor(p0, off, 32);
            p1 += __shfl_xor(p1, off, 32);
        }
        if (lane == 0) {
            out[row * 2 + 0] = p0 + b2[0];
            out[row * 2 + 1] = p1 + b2[1];
        }
    }
}

extern "C" void kernel_launch(void* const* d_in, const int* in_sizes, int n_in,
                              void* d_out, int out_size, void* d_ws, size_t ws_size,
                              hipStream_t stream) {
    const float* h      = (const float*)d_in[0];
    const int*   esrc   = (const int*)d_in[1];
    const int*   edst   = (const int*)d_in[2];
    const int*   x1     = (const int*)d_in[3];
    const int*   x2     = (const int*)d_in[4];
    const int*   x1t    = (const int*)d_in[5];
    const int*   x2t    = (const int*)d_in[6];
    const float* Wself  = (const float*)d_in[7];   // [2,128,128]
    const float* Wneigh = (const float*)d_in[8];   // [2,128,128]
    const float* bconv  = (const float*)d_in[9];   // [2,128]
    const float* W1     = (const float*)d_in[10];  // [384,128]
    const float* b1     = (const float*)d_in[11];
    const float* gamma  = (const float*)d_in[12];
    const float* beta   = (const float*)d_in[13];
    const float* W2     = (const float*)d_in[14];  // [128,2]
    const float* b2     = (const float*)d_in[15];

    const int N = in_sizes[0] / 128;
    const int E = in_sizes[1];
    const int B = in_sizes[3];
    const int ND = N * 128;

    // ---- workspace carve-up (floats) ----
    float* ws = (float*)d_ws;
    size_t o = 0;
    float* deg  = ws + o; o += (size_t)((N + 63) / 64) * 64;
    float* agg  = ws + o; o += (size_t)ND;
    float* hA   = ws + o; o += (size_t)ND;
    float* hB   = ws + o; o += (size_t)ND;
    float* zsrc = ws + o; o += (size_t)B * 128;
    float* ztar = ws + o; o += (size_t)B * 128;
    float* stats= ws + o; o += 512;
    unsigned short* wp = (unsigned short*)(ws + o);
    o += (115000 + 1) / 2;                  // 114688 halves of packed weights
    unsigned short* WpS0 = wp;              // 128*128 halves each
    unsigned short* WpN0 = wp + 16384;
    unsigned short* WpS1 = wp + 32768;
    unsigned short* WpN1 = wp + 49152;
    unsigned short* Wp1  = wp + 65536;      // 384*128 halves
    // packed A buffers: reused as (h, agg) in conv layers, then (src, tar)
    // pair-feature packs in the head (B*384 <= N*128 halves each).
    unsigned short* ApX = (unsigned short*)(ws + o); o += (size_t)ND / 2;
    unsigned short* ApY = (unsigned short*)(ws + o); o += (size_t)ND / 2;

    // ---- pack weights into WMMA B-operand layout (bf16) ----
    pack_w_kernel<<<64, 256, 0, stream>>>(Wself,          WpS0, 128);
    pack_w_kernel<<<64, 256, 0, stream>>>(Wself + 16384,  WpS1, 128);
    pack_w_kernel<<<64, 256, 0, stream>>>(Wneigh,         WpN0, 128);
    pack_w_kernel<<<64, 256, 0, stream>>>(Wneigh + 16384, WpN1, 128);
    pack_w_kernel<<<192, 256, 0, stream>>>(W1,            Wp1,  384);

    // ---- degree / reciprocal ----
    zero_kernel<<<(N + 255) / 256, 256, 0, stream>>>(deg, N);
    deg_kernel<<<(E + 255) / 256, 256, 0, stream>>>(edst, deg, E);
    deginv_kernel<<<(N + 255) / 256, 256, 0, stream>>>(deg, N);

    const int Mtiles = N / 16;              // 3125
    const int Btiles = B / 16;              // 1024
    const int sageStrips = Mtiles / 5;      // 3125 = 5 * 625
    const int pairStrips = Btiles / 4;      // 1024 = 4 * 256
    const int scatterBlocks = (E * 32 + 255) / 256;
    const int aggBlocks = (ND + 255) / 256;
    const int packABlocks = Mtiles;         // Mtiles*256 threads
    const int sageBlocks = (sageStrips * 8 + 7) / 8;
    const int pairBlocks = (pairStrips * 8 + 7) / 8;
    const int packPairBlocks = (Btiles * 768 + 255) / 256;

    // ---- layer 0 ----
    zero_kernel<<<aggBlocks, 256, 0, stream>>>(agg, ND);
    scatter_kernel<<<scatterBlocks, 256, 0, stream>>>(h, esrc, edst, agg, E);
    pack_a_kernel<<<packABlocks, 256, 0, stream>>>(h,   nullptr, ApX, Mtiles);
    pack_a_kernel<<<packABlocks, 256, 0, stream>>>(agg, deg,     ApY, Mtiles);
    wmma_gemm_kernel<4, 5, true, true><<<sageBlocks, 256, 0, stream>>>(
        ApX, ApY, WpS0, WpN0, bconv, hA, sageStrips);

    // ---- layer 1 ----
    zero_kernel<<<aggBlocks, 256, 0, stream>>>(agg, ND);
    scatter_kernel<<<scatterBlocks, 256, 0, stream>>>(hA, esrc, edst, agg, E);
    pack_a_kernel<<<packABlocks, 256, 0, stream>>>(hA,  nullptr, ApX, Mtiles);
    pack_a_kernel<<<packABlocks, 256, 0, stream>>>(agg, deg,     ApY, Mtiles);
    wmma_gemm_kernel<4, 5, true, true><<<sageBlocks, 256, 0, stream>>>(
        ApX, ApY, WpS1, WpN1, bconv + 128, hB, sageStrips);

    // ---- pair head: pack gathered features once, then pure-WMMA GEMMs ----
    pack_pair_kernel<<<packPairBlocks, 256, 0, stream>>>(hB, x1,  x2,  ApX, Btiles);
    pack_pair_kernel<<<packPairBlocks, 256, 0, stream>>>(hB, x1t, x2t, ApY, Btiles);
    wmma_gemm_kernel<12, 4, false, false><<<pairBlocks, 256, 0, stream>>>(
        ApX, nullptr, Wp1, nullptr, b1, zsrc, pairStrips);
    wmma_gemm_kernel<12, 4, false, false><<<pairBlocks, 256, 0, stream>>>(
        ApY, nullptr, Wp1, nullptr, b1, ztar, pairStrips);

    // ---- batchnorm stats + fused finalize ----
    bn_stats_kernel<<<256, 256, 0, stream>>>(zsrc, ztar, stats, B);
    finalize_kernel<<<(2 * B + 7) / 8, 256, 0, stream>>>(zsrc, ztar, stats, gamma,
                                                         beta, W2, b2,
                                                         (float*)d_out, B);
}
